// GAUEncoder_38448547234725
// MI455X (gfx1250) — matrix operand order repaired and checked by reading, not compiled, not run on hardware
//
#include <hip/hip_runtime.h>

// GAU encoder on MI455X (gfx1250, wave32). All GEMMs via
// v_wmma_f32_16x16x32_bf16, double-buffered LDS staging with async
// global->LDS loads (ASYNCcnt) when the toolchain exposes them.

#define Lc   4
#define Bc   4
#define Sc   2048
#define DMc  1024
#define DFFc 2048
#define HSc  128
#define BSc  (Bc * Sc)

typedef __attribute__((ext_vector_type(16))) __bf16 v16bf;
typedef __attribute__((ext_vector_type(2)))  __bf16 bf16x2;
typedef __attribute__((ext_vector_type(8)))  float  v8f;

static __device__ __forceinline__ __bf16 f2bf(float f) { return (__bf16)f; }

#if defined(__HIP_DEVICE_COMPILE__) && __has_builtin(__builtin_amdgcn_global_load_async_to_lds_b64)
#define HAS_ASYNC 1
#else
#define HAS_ASYNC 0
#endif

#if HAS_ASYNC
// Builtin signature (from compiler diagnostic): arg0 is a pointer to
// int __attribute__((vector_size(8))) in the global (__device__/as1) address
// space; arg1 is the LDS (as3) pointer; args 2,3 are imm offset / cpol.
typedef int v2i_t __attribute__((vector_size(8)));
typedef __attribute__((address_space(1))) v2i_t gv2i_t;
typedef __attribute__((address_space(3))) v2i_t lv2i_t;

static __device__ __forceinline__ void async_ld_b64(const __bf16* g, __bf16* l) {
    __builtin_amdgcn_global_load_async_to_lds_b64(
        (gv2i_t*)(uintptr_t)g,
        (lv2i_t*)(unsigned int)(uintptr_t)l, 0, 0);
}
#endif

static __device__ __forceinline__ void wait_async0() {
#if HAS_ASYNC
# if __has_builtin(__builtin_amdgcn_s_wait_asynccnt)
    __builtin_amdgcn_s_wait_asynccnt(0);
# else
    asm volatile("s_wait_asynccnt 0x0" ::: "memory");
# endif
#endif
}

// ---------------------------------------------------------------------------
// f32 -> bf16 (activations)
// ---------------------------------------------------------------------------
__global__ void k_cvt_bf16(const float* __restrict__ in, __bf16* __restrict__ out, int n) {
    int i = blockIdx.x * blockDim.x + threadIdx.x;
    if (i < n) out[i] = f2bf(in[i]);
}

// ---------------------------------------------------------------------------
// f32 [K,N] -> bf16 [N,K] tiled transpose (weights, once per call)
// ---------------------------------------------------------------------------
__global__ __launch_bounds__(256)
void k_cvt_transpose(const float* __restrict__ in, __bf16* __restrict__ out, int K, int N) {
    __shared__ float t[32][33];
    const size_t slice = (size_t)K * N;
    const float* src = in + (size_t)blockIdx.z * slice;
    __bf16*      dst = out + (size_t)blockIdx.z * slice;
    int n0 = blockIdx.x * 32, k0 = blockIdx.y * 32;
    int tx = threadIdx.x & 31, ty = threadIdx.x >> 5;
    #pragma unroll
    for (int i = 0; i < 4; ++i) {
        int k = ty + i * 8;
        t[k][tx] = src[(size_t)(k0 + k) * N + n0 + tx];
    }
    __syncthreads();
    #pragma unroll
    for (int i = 0; i < 4; ++i) {
        int n = ty + i * 8;
        dst[(size_t)(n0 + n) * K + k0 + tx] = f2bf(t[tx][n]);
    }
}

// ---------------------------------------------------------------------------
// Tiled bf16 WMMA GEMM: C[M,N] = A[M,K] @ B' where
//   BT=1: B is [N,K] row-major (direct, async-friendly staging)
//   BT=0: B is [K,N] row-major (transposed while staging)
// Block tile 128x128, BK=32, double-buffered LDS. 8 waves, each 32x64
// (2x4 WMMA tiles -> 8 v_wmma per K-step per wave).
// EPI: 0 f32 store | 1 bf16 store | 2 relu^2*scale*mask->bf16 |
//      3 *u gate->bf16 | 4 +residual->f32
// ---------------------------------------------------------------------------
template<int BT, int EPI>
__global__ __launch_bounds__(256)
void k_gemm(const __bf16* __restrict__ A, unsigned long long strideA, int lda,
            const __bf16* __restrict__ B, unsigned long long strideB, int ldb,
            int M, int N, int K,
            void* __restrict__ C, unsigned long long strideC,
            const void* __restrict__ aux, unsigned long long strideAux,
            float scale) {
    // 36-element (72B) pitch: padded against bank conflicts for the ISA
    // 16-bit A/B fragment access patterns on 64 banks; 8B-aligned rows
    // keep async b64 LDS writes legal.
    __shared__ __bf16 As[2][128 * 36];
    __shared__ __bf16 Bs[2][128 * 36];

    const int tid    = threadIdx.x;
    const int lane   = tid & 31;
    const int wave   = tid >> 5;
    const int waveM  = (wave >> 1) * 32;    // 0,32,64,96
    const int waveN  = (wave & 1) * 64;     // 0,64
    const int batch  = blockIdx.z;
    const int blockM = blockIdx.y * 128;
    const int blockN = blockIdx.x * 128;

    const __bf16* Ab = A + (unsigned long long)batch * strideA;
    const __bf16* Bb = B + (unsigned long long)batch * strideB;

    const v8f vzero = {0.f, 0.f, 0.f, 0.f, 0.f, 0.f, 0.f, 0.f};
    v8f acc[2][4];
    #pragma unroll
    for (int mt = 0; mt < 2; ++mt)
        #pragma unroll
        for (int nt = 0; nt < 4; ++nt) acc[mt][nt] = vzero;

    const int mrow  = lane & 15;
    const int halfl = lane >> 4;

    auto stage = [&](int k0, int buf) {
        // ---- A tile 128x32 ----
#if HAS_ASYNC
        #pragma unroll
        for (int i = 0; i < 4; ++i) {
            int id = tid + i * 256;
            int r = id >> 3, cp = id & 7;            // 8B chunks
            async_ld_b64(Ab + (unsigned long long)(blockM + r) * lda + k0 + cp * 4,
                         &As[buf][r * 36 + cp * 4]);
        }
#else
        #pragma unroll
        for (int i = 0; i < 2; ++i) {
            int id = tid + i * 256;
            int r = id >> 2, c = id & 3;             // 16B chunks
            const uint4* src = reinterpret_cast<const uint4*>(
                Ab + (unsigned long long)(blockM + r) * lda + k0 + c * 8);
            uint4 d = *src;
            unsigned int* dst = reinterpret_cast<unsigned int*>(&As[buf][r * 36 + c * 8]);
            dst[0] = d.x; dst[1] = d.y; dst[2] = d.z; dst[3] = d.w;
        }
#endif
        // ---- B tile -> Bs[n][k] ----
        if (BT) {
#if HAS_ASYNC
            #pragma unroll
            for (int i = 0; i < 4; ++i) {
                int id = tid + i * 256;
                int r = id >> 3, cp = id & 7;
                async_ld_b64(Bb + (unsigned long long)(blockN + r) * ldb + k0 + cp * 4,
                             &Bs[buf][r * 36 + cp * 4]);
            }
#else
            #pragma unroll
            for (int i = 0; i < 2; ++i) {
                int id = tid + i * 256;
                int r = id >> 2, c = id & 3;
                const uint4* src = reinterpret_cast<const uint4*>(
                    Bb + (unsigned long long)(blockN + r) * ldb + k0 + c * 8);
                uint4 d = *src;
                unsigned int* dst = reinterpret_cast<unsigned int*>(&Bs[buf][r * 36 + c * 8]);
                dst[0] = d.x; dst[1] = d.y; dst[2] = d.z; dst[3] = d.w;
            }
#endif
        } else {
            // transpose while staging: 2x2 pair-merge, b32 LDS stores
            #pragma unroll
            for (int i = 0; i < 4; ++i) {
                int id = tid + i * 256;
                int kp = id & 15;                    // k pair: 2*kp
                int np = id >> 4;                    // n pair: 2*np (0..63)
                int k = 2 * kp, n = 2 * np;
                unsigned int u0 = *reinterpret_cast<const unsigned int*>(
                    Bb + (unsigned long long)(k0 + k) * ldb + blockN + n);
                unsigned int u1 = *reinterpret_cast<const unsigned int*>(
                    Bb + (unsigned long long)(k0 + k + 1) * ldb + blockN + n);
                unsigned int w0 = (u0 & 0xffffu) | (u1 << 16);
                unsigned int w1 = (u0 >> 16) | (u1 & 0xffff0000u);
                *reinterpret_cast<unsigned int*>(&Bs[buf][(n + 0) * 36 + k]) = w0;
                *reinterpret_cast<unsigned int*>(&Bs[buf][(n + 1) * 36 + k]) = w1;
            }
        }
    };

    const int nk = K >> 5;
    stage(0, 0);
    int buf = 0;
    for (int it = 0; it < nk; ++it) {
        wait_async0();
        __syncthreads();
        if (it + 1 < nk) stage((it + 1) << 5, buf ^ 1);

        v16bf af[2], bfr[4];
        #pragma unroll
        for (int mt = 0; mt < 2; ++mt) {
            int mb = waveM + mt * 16 + mrow;
            #pragma unroll
            for (int r = 0; r < 8; ++r) {
                int kk = ((r < 4) ? (2 * r) : (16 + 2 * (r - 4))) + halfl * 8;
                bf16x2 p = *reinterpret_cast<const bf16x2*>(&As[buf][mb * 36 + kk]);
                af[mt][2 * r]     = p[0];
                af[mt][2 * r + 1] = p[1];
            }
        }
        #pragma unroll
        for (int nt = 0; nt < 4; ++nt) {
            int nb = waveN + nt * 16 + mrow;
            #pragma unroll
            for (int r = 0; r < 8; ++r) {
                int kk = ((r < 4) ? (2 * r) : (16 + 2 * (r - 4))) + halfl * 8;
                bf16x2 p = *reinterpret_cast<const bf16x2*>(&Bs[buf][nb * 36 + kk]);
                bfr[nt][2 * r]     = p[0];
                bfr[nt][2 * r + 1] = p[1];
            }
        }
        #pragma unroll
        for (int mt = 0; mt < 2; ++mt)
            #pragma unroll
            for (int nt = 0; nt < 4; ++nt)
                acc[mt][nt] = __builtin_amdgcn_wmma_f32_16x16x32_bf16(
                    false, af[mt], false, bfr[nt], (short)0, acc[mt][nt], false, false);
        buf ^= 1;
    }

    // epilogue: C f32 layout row = half*8 + r, col = lane&15
    #pragma unroll
    for (int mt = 0; mt < 2; ++mt) {
        #pragma unroll
        for (int nt = 0; nt < 4; ++nt) {
            #pragma unroll
            for (int r = 0; r < 8; ++r) {
                int row = blockM + waveM + mt * 16 + halfl * 8 + r;
                int col = blockN + waveN + nt * 16 + (lane & 15);
                float v = acc[mt][nt][r];
                unsigned long long gi = (unsigned long long)row * N + col;
                unsigned long long ci = (unsigned long long)batch * strideC + gi;
                unsigned long long xi = (unsigned long long)batch * strideAux + gi;
                if (EPI == 0) {
                    reinterpret_cast<float*>(C)[ci] = v;
                } else if (EPI == 1) {
                    reinterpret_cast<__bf16*>(C)[ci] = f2bf(v);
                } else if (EPI == 2) {
                    float s = fmaxf(v, 0.f);
                    s = s * s * scale;
                    if (!reinterpret_cast<const unsigned char*>(aux)[xi]) s = 0.f;
                    reinterpret_cast<__bf16*>(C)[ci] = f2bf(s);
                } else if (EPI == 3) {
                    float u = reinterpret_cast<const float*>(aux)[xi];
                    reinterpret_cast<__bf16*>(C)[ci] = f2bf(v * u);
                } else {
                    float h = reinterpret_cast<const float*>(aux)[xi];
                    reinterpret_cast<float*>(C)[ci] = v + h;
                }
            }
        }
    }
}

// ---------------------------------------------------------------------------
// RoPE: bug-faithful stack((-odd, even)).reshape flat-index semantics
// ---------------------------------------------------------------------------
__global__ void k_rope_qk(const float* __restrict__ z,
                          const float* __restrict__ gq, const float* __restrict__ bq,
                          const float* __restrict__ gk, const float* __restrict__ bk,
                          __bf16* __restrict__ qo, __bf16* __restrict__ ko) {
    const int half = HSc / 2;
    unsigned int tid = blockIdx.x * blockDim.x + threadIdx.x;
    unsigned int sh  = tid % (Sc * HSc);
    int s = sh / HSc;
    int e = sh % HSc;
    int i = e >> 1;
    float freq = __expf(-(float)i * (9.210340371976184f / (float)half));
    float ang  = (float)s * freq;
    float sn = __sinf(ang), cs = __cosf(ang);

    float zx = z[tid];
    float qx = zx * gq[e] + bq[e];
    float kx = zx * gk[e] + bk[e];

    unsigned int blk = (unsigned int)Bc * Sc * half;
    unsigned int d   = tid / blk;
    unsigned int rr  = tid % blk;
    unsigned int b2  = rr / (Sc * half);
    unsigned int r2  = rr % (Sc * half);
    unsigned int s2  = r2 / half;
    unsigned int j   = r2 % half;
    unsigned int base = (b2 * Sc + s2) * HSc + 2u * j;
    float z_ev = z[base], z_od = z[base + 1];
    int e_ev = 2 * j, e_od = 2 * j + 1;

    float qh2 = (d == 0) ? -(z_od * gq[e_od] + bq[e_od]) : (z_ev * gq[e_ev] + bq[e_ev]);
    float kh2 = (d == 0) ? -(z_od * gk[e_od] + bk[e_od]) : (z_ev * gk[e_ev] + bk[e_ev]);

    qo[tid] = f2bf(qx * cs + qh2 * sn);
    ko[tid] = f2bf(kx * cs + kh2 * sn);
}

// ---------------------------------------------------------------------------
// RMSNorm over DM=1024
// ---------------------------------------------------------------------------
__global__ __launch_bounds__(256)
void k_rmsnorm(const float* __restrict__ x, const float* __restrict__ nw,
               float* __restrict__ y) {
    __shared__ float red[256];
    const int row = blockIdx.x;
    const int t   = threadIdx.x;
    const float* xr = x + (unsigned long long)row * DMc;
    float v0 = xr[t], v1 = xr[t + 256], v2 = xr[t + 512], v3 = xr[t + 768];
    red[t] = v0 * v0 + v1 * v1 + v2 * v2 + v3 * v3;
    __syncthreads();
    for (int s = 128; s > 0; s >>= 1) {
        if (t < s) red[t] += red[t + s];
        __syncthreads();
    }
    float rs = rsqrtf(red[0] / (float)DMc + 1e-5f);
    float* yr = y + (unsigned long long)row * DMc;
    yr[t]       = v0 * rs * nw[t];
    yr[t + 256] = v1 * rs * nw[t + 256];
    yr[t + 512] = v2 * rs * nw[t + 512];
    yr[t + 768] = v3 * rs * nw[t + 768];
}

// ---------------------------------------------------------------------------
// host driver
// ---------------------------------------------------------------------------
extern "C" void kernel_launch(void* const* d_in, const int* in_sizes, int n_in,
                              void* d_out, int out_size, void* d_ws, size_t ws_size,
                              hipStream_t stream) {
    const float* hs = (const float*)d_in[0];
    const unsigned char* mask = (const unsigned char*)d_in[1];
    const float* Wu = (const float*)d_in[2];
    const float* Wv = (const float*)d_in[3];
    const float* Wh = (const float*)d_in[4];
    const float* gq = (const float*)d_in[5];
    const float* bq = (const float*)d_in[6];
    const float* gk = (const float*)d_in[7];
    const float* bk = (const float*)d_in[8];
    const float* Wb = (const float*)d_in[9];
    const float* nw = (const float*)d_in[10];

    char*  base = (char*)d_ws;
    size_t off  = 0;
    auto alloc = [&](size_t bytes) -> char* {
        char* p = base + off;
        off += (bytes + 255) & ~(size_t)255;
        return p;
    };
    __bf16* wuT  = (__bf16*)alloc((size_t)Lc * DMc * DFFc * 2);   // [L,DFF,DM]
    __bf16* wvT  = (__bf16*)alloc((size_t)Lc * DMc * DFFc * 2);   // [L,DFF,DM]
    __bf16* whT  = (__bf16*)alloc((size_t)Lc * DMc * HSc * 2);    // [L,HS,DM]
    __bf16* wbT  = (__bf16*)alloc((size_t)Lc * DFFc * DMc * 2);   // [L,DM,DFF]
    __bf16* hbf  = (__bf16*)alloc((size_t)BSc * DMc * 2);
    float*  hA   = (float*) alloc((size_t)BSc * DMc * 4);
    float*  hB   = (float*) alloc((size_t)BSc * DMc * 4);
    float*  ubuf = (float*) alloc((size_t)BSc * DFFc * 4);        // reused as pre-norm
    __bf16* vbf  = (__bf16*)alloc((size_t)BSc * DFFc * 2);
    float*  zbuf = (float*) alloc((size_t)BSc * HSc * 4);
    __bf16* qbf  = (__bf16*)alloc((size_t)BSc * HSc * 2);
    __bf16* kbf  = (__bf16*)alloc((size_t)BSc * HSc * 2);
    __bf16* sbf  = (__bf16*)alloc((size_t)Bc * Sc * Sc * 2);
    __bf16* gbf  = (__bf16*)alloc((size_t)BSc * DFFc * 2);

    const float scale = 1.0f / ((float)Sc * (float)HSc);

    // weights -> bf16, transposed to [N,K]
    k_cvt_transpose<<<dim3(DFFc / 32, DMc / 32, Lc), 256, 0, stream>>>(Wu, wuT, DMc, DFFc);
    k_cvt_transpose<<<dim3(DFFc / 32, DMc / 32, Lc), 256, 0, stream>>>(Wv, wvT, DMc, DFFc);
    k_cvt_transpose<<<dim3(HSc / 32,  DMc / 32, Lc), 256, 0, stream>>>(Wh, whT, DMc, HSc);
    k_cvt_transpose<<<dim3(DMc / 32,  DFFc / 32, Lc), 256, 0, stream>>>(Wb, wbT, DFFc, DMc);

    const float* hin = hs;
    for (int l = 0; l < Lc; ++l) {
        float* hout = (l == Lc - 1) ? (float*)d_out : ((l & 1) ? hB : hA);

        int nh = BSc * DMc;
        k_cvt_bf16<<<nh / 256, 256, 0, stream>>>(hin, hbf, nh);

        // u = h @ Wu (f32)
        k_gemm<1, 0><<<dim3(DFFc / 128, BSc / 128, 1), 256, 0, stream>>>(
            hbf, 0ull, DMc, wuT + (size_t)l * DMc * DFFc, 0ull, DMc,
            BSc, DFFc, DMc, ubuf, 0ull, nullptr, 0ull, 0.f);
        // v = h @ Wv (bf16)
        k_gemm<1, 1><<<dim3(DFFc / 128, BSc / 128, 1), 256, 0, stream>>>(
            hbf, 0ull, DMc, wvT + (size_t)l * DMc * DFFc, 0ull, DMc,
            BSc, DFFc, DMc, vbf, 0ull, nullptr, 0ull, 0.f);
        // z = h @ Wh (f32)
        k_gemm<1, 0><<<dim3(HSc / 128, BSc / 128, 1), 256, 0, stream>>>(
            hbf, 0ull, DMc, whT + (size_t)l * DMc * HSc, 0ull, DMc,
            BSc, HSc, DMc, zbuf, 0ull, nullptr, 0ull, 0.f);

        k_rope_qk<<<(BSc * HSc) / 256, 256, 0, stream>>>(
            zbuf, gq + l * HSc, bq + l * HSc, gk + l * HSc, bk + l * HSc, qbf, kbf);

        // score = mask(relu(q @ k^T)^2 * scale) (bf16, batched)
        k_gemm<1, 2><<<dim3(Sc / 128, Sc / 128, Bc), 256, 0, stream>>>(
            qbf, (unsigned long long)Sc * HSc, HSc,
            kbf, (unsigned long long)Sc * HSc, HSc,
            Sc, Sc, HSc, sbf, (unsigned long long)Sc * Sc,
            mask, (unsigned long long)Sc * Sc, scale);

        // gau = u * (score @ v) (bf16, batched; B=[K,N] transposed staging)
        k_gemm<0, 3><<<dim3(DFFc / 128, Sc / 128, Bc), 256, 0, stream>>>(
            sbf, (unsigned long long)Sc * Sc, Sc,
            vbf, (unsigned long long)Sc * DFFc, DFFc,
            Sc, DFFc, Sc, gbf, (unsigned long long)Sc * DFFc,
            ubuf, (unsigned long long)Sc * DFFc, 0.f);

        // pre = gau @ Wb + h (f32, reuse ubuf)
        k_gemm<1, 4><<<dim3(DMc / 128, BSc / 128, 1), 256, 0, stream>>>(
            gbf, 0ull, DFFc, wbT + (size_t)l * DFFc * DMc, 0ull, DFFc,
            BSc, DMc, DFFc, ubuf, 0ull, hin, 0ull, 0.f);

        k_rmsnorm<<<BSc, 256, 0, stream>>>(ubuf, nw + l * DMc, hout);

        hin = hout;
    }
    (void)in_sizes; (void)n_in; (void)out_size; (void)ws_size;
}
